// CorefLongformerModel_9448928051517
// MI455X (gfx1250) — compile-verified, matrix operand order checked by reading, not run on hardware
//
#include <hip/hip_runtime.h>
#include <hip/hip_bf16.h>
#include <math.h>

// ---------------- model constants ----------------
#define B_    2
#define L_    4096
#define D_    768
#define H_    12
#define DH_   64
#define W_    256
#define G_    16
#define FF_   3072
#define NLAB_ 3
#define LEMB_ 10
#define GRUH_ 768
#define NC_   (L_ / W_)        // 16 chunks
#define NEGV  (-1.0e9f)
#define GIK_  800              // 778 padded to multiple of 32

// GEMM blocking
#define BM_   128
#define BN_   128
#define LDS_K 40               // 32 + 8 pad halves -> conflict-free b128 reads

typedef __attribute__((ext_vector_type(16))) _Float16 v16h;
typedef __attribute__((ext_vector_type(8)))  _Float16 v8h;
typedef __attribute__((ext_vector_type(8)))  float    v8f;

// ---------------- WMMA helpers ----------------
__device__ inline v8f wmma_f16(v16h a, v16h b, v8f c) {
  return __builtin_amdgcn_wmma_f32_16x16x32_f16(false, a, false, b, (short)0, c,
                                                false, false);
}

// A/B fragment (16x32 f16, row-major source [rows, ld]):
// lane m = row0 + (lane&15); elems 0..7 <- K=k0+hi*8+0..7 ; 8..15 <- K=k0+16+hi*8+0..7
__device__ inline v16h frag_load_rowmajor(const _Float16* __restrict__ base, int ld,
                                          int row0, int k0) {
  int lane = threadIdx.x & 31;
  int hi   = lane >> 4;
  const _Float16* p = base + (size_t)(row0 + (lane & 15)) * ld + k0 + hi * 8;
  v8h lo = *(const v8h*)(p);
  v8h hh = *(const v8h*)(p + 16);
  v16h f;
#pragma unroll
  for (int i = 0; i < 8; ++i) { f[i] = lo[i]; f[i + 8] = hh[i]; }
  return f;
}

// fragment from LDS panel [rows][LDS_K]
__device__ inline v16h frag_load_lds(const _Float16* base, int row0) {
  int lane = threadIdx.x & 31;
  int hi   = lane >> 4;
  const _Float16* p = base + (row0 + (lane & 15)) * LDS_K + hi * 8;
  v8h lo = *(const v8h*)(p);
  v8h hh = *(const v8h*)(p + 16);
  v16h f;
#pragma unroll
  for (int i = 0; i < 8; ++i) { f[i] = lo[i]; f[i + 8] = hh[i]; }
  return f;
}

// B fragment where the per-lane row index ("key") may be out of range -> zeros
__device__ inline v16h frag_load_rowguard(const _Float16* __restrict__ base, int ld,
                                          int key, int keymax, int k0) {
  int lane = threadIdx.x & 31;
  int hi   = lane >> 4;
  v16h f;
  if (key >= 0 && key < keymax) {
    const _Float16* p = base + (size_t)key * ld + k0 + hi * 8;
    v8h lo = *(const v8h*)(p);
    v8h hh = *(const v8h*)(p + 16);
#pragma unroll
    for (int i = 0; i < 8; ++i) { f[i] = lo[i]; f[i + 8] = hh[i]; }
  } else {
#pragma unroll
    for (int i = 0; i < 16; ++i) f[i] = (_Float16)0.f;
  }
  return f;
}

// B fragment for transposed-V: row n = n0+(lane&15) of Vt[ld_rows, kmax],
// column (K) runs guarded against [0,kmax). Runs are 8-aligned so run-level guard ok.
__device__ inline v16h frag_load_colguard(const _Float16* __restrict__ base, int ld,
                                          int n0, int kb, int kmax) {
  int lane = threadIdx.x & 31;
  int hi   = lane >> 4;
  const _Float16* row = base + (size_t)(n0 + (lane & 15)) * ld;
  int k0 = kb + hi * 8;
  int k1 = kb + 16 + hi * 8;
  v16h f;
  if (k0 >= 0 && k0 < kmax) {
    v8h t = *(const v8h*)(row + k0);
#pragma unroll
    for (int i = 0; i < 8; ++i) f[i] = t[i];
  } else {
#pragma unroll
    for (int i = 0; i < 8; ++i) f[i] = (_Float16)0.f;
  }
  if (k1 >= 0 && k1 < kmax) {
    v8h t = *(const v8h*)(row + k1);
#pragma unroll
    for (int i = 0; i < 8; ++i) f[i + 8] = t[i];
  } else {
#pragma unroll
    for (int i = 0; i < 8; ++i) f[i + 8] = (_Float16)0.f;
  }
  return f;
}

// ---------------- LDS-blocked WMMA GEMM ----------------
// C[M,N] = A[M,K](f16) * Bt[N,K](f16)^T + bias ; act: 0 none, 1 gelu(tanh)
// Block tile 128x128, 8 waves (4m x 2n), each wave 32x64 = 2x4 WMMA tiles.
// Double-buffered LDS panels; one barrier per K-step.
__global__ void __launch_bounds__(256)
k_gemm(const _Float16* __restrict__ A, const _Float16* __restrict__ Bt,
       const float* __restrict__ bias, float* __restrict__ Cf32,
       _Float16* __restrict__ Cf16, int M, int N, int K, int act) {
  __shared__ _Float16 As[2][BM_ * LDS_K];
  __shared__ _Float16 Bs[2][BN_ * LDS_K];
  int nbn = N / BN_;
  int bm = blockIdx.x / nbn;
  int bn = blockIdx.x % nbn;
  int tid  = threadIdx.x;
  int wave = tid >> 5;
  int lane = tid & 31;
  int lan = lane & 15, hi = lane >> 4;
  int wm = wave >> 1;            // 0..3 : 32-row strip
  int wn = wave & 1;             // 0..1 : 64-col strip

  // cooperative load mapping: 512 chunks of 8 halves per panel, 2 per thread
  int crow = tid >> 2;           // 0..63
  int ck   = (tid & 3) * 8;      // 0,8,16,24
  const _Float16* Abase = A  + (size_t)(bm * BM_) * K;
  const _Float16* Bbase = Bt + (size_t)(bn * BN_) * K;

  v8f acc[2][4];
#pragma unroll
  for (int i = 0; i < 2; ++i)
#pragma unroll
    for (int j = 0; j < 4; ++j) acc[i][j] = (v8f){};

  int steps = K >> 5;
  v8h ra0, ra1, rb0, rb1;
  // prologue fetch (k-step 0)
  ra0 = *(const v8h*)(Abase + (size_t)crow * K + ck);
  ra1 = *(const v8h*)(Abase + (size_t)(crow + 64) * K + ck);
  rb0 = *(const v8h*)(Bbase + (size_t)crow * K + ck);
  rb1 = *(const v8h*)(Bbase + (size_t)(crow + 64) * K + ck);

  for (int kt = 0; kt < steps; ++kt) {
    int buf = kt & 1;
    *(v8h*)(&As[buf][crow * LDS_K + ck])        = ra0;
    *(v8h*)(&As[buf][(crow + 64) * LDS_K + ck]) = ra1;
    *(v8h*)(&Bs[buf][crow * LDS_K + ck])        = rb0;
    *(v8h*)(&Bs[buf][(crow + 64) * LDS_K + ck]) = rb1;
    __syncthreads();
    if (kt + 1 < steps) {
      int k0 = (kt + 1) << 5;
      ra0 = *(const v8h*)(Abase + (size_t)crow * K + k0 + ck);
      ra1 = *(const v8h*)(Abase + (size_t)(crow + 64) * K + k0 + ck);
      rb0 = *(const v8h*)(Bbase + (size_t)crow * K + k0 + ck);
      rb1 = *(const v8h*)(Bbase + (size_t)(crow + 64) * K + k0 + ck);
      if (kt + 2 < steps) {
        __builtin_prefetch(Abase + (size_t)crow * K + k0 + 32 + ck, 0, 1);
        __builtin_prefetch(Bbase + (size_t)crow * K + k0 + 32 + ck, 0, 1);
      }
    }
    v16h af0 = frag_load_lds(&As[buf][0], wm * 32);
    v16h af1 = frag_load_lds(&As[buf][0], wm * 32 + 16);
#pragma unroll
    for (int j = 0; j < 4; ++j) {
      v16h bf = frag_load_lds(&Bs[buf][0], wn * 64 + j * 16);
      acc[0][j] = wmma_f16(af0, bf, acc[0][j]);
      acc[1][j] = wmma_f16(af1, bf, acc[1][j]);
    }
  }

  // epilogue
#pragma unroll
  for (int i = 0; i < 2; ++i) {
#pragma unroll
    for (int j = 0; j < 4; ++j) {
      int n = bn * BN_ + wn * 64 + j * 16 + lan;
      float bv = bias ? bias[n] : 0.f;
#pragma unroll
      for (int r = 0; r < 8; ++r) {
        int m = bm * BM_ + wm * 32 + i * 16 + r + 8 * hi;
        float v = acc[i][j][r] + bv;
        if (act == 1) {
          float x3 = v * v * v;
          v = 0.5f * v * (1.f + tanhf(0.7978845608028654f * (v + 0.044715f * x3)));
        }
        if (Cf32) Cf32[(size_t)m * N + n] = v;
        if (Cf16) Cf16[(size_t)m * N + n] = (_Float16)v;
      }
    }
  }
}

// ---------------- elementwise / conversion kernels ----------------
__global__ void k_embed(const int* __restrict__ tok, const float* __restrict__ temb,
                        const float* __restrict__ pemb, float* __restrict__ x) {
  size_t i = (size_t)blockIdx.x * blockDim.x + threadIdx.x;
  if (i >= (size_t)B_ * L_ * D_) return;
  int d = (int)(i % D_);
  size_t bl = i / D_;
  int l = (int)(bl % L_);
  int t = tok[bl];
  x[i] = temb[(size_t)t * D_ + d] + pemb[(size_t)l * D_ + d];
}

// LayerNorm over last dim (ncols); optional residual; writes f32 and f16 views
__global__ void k_ln(const float* __restrict__ xin, const float* __restrict__ res,
                     const float* __restrict__ g, const float* __restrict__ b,
                     float* __restrict__ o32, _Float16* __restrict__ o16, int ncols) {
  __shared__ float red[256];
  size_t row = blockIdx.x;
  const float* px = xin + row * ncols;
  const float* pr = res ? res + row * ncols : nullptr;
  int tid = threadIdx.x;
  float s = 0.f;
  for (int c = tid; c < ncols; c += blockDim.x) s += px[c] + (pr ? pr[c] : 0.f);
  red[tid] = s; __syncthreads();
  for (int st = 128; st > 0; st >>= 1) { if (tid < st) red[tid] += red[tid + st]; __syncthreads(); }
  float mu = red[0] / ncols; __syncthreads();
  float vs = 0.f;
  for (int c = tid; c < ncols; c += blockDim.x) {
    float v = px[c] + (pr ? pr[c] : 0.f) - mu; vs += v * v;
  }
  red[tid] = vs; __syncthreads();
  for (int st = 128; st > 0; st >>= 1) { if (tid < st) red[tid] += red[tid + st]; __syncthreads(); }
  float inv = rsqrtf(red[0] / ncols + 1e-5f);
  for (int c = tid; c < ncols; c += blockDim.x) {
    float v = (px[c] + (pr ? pr[c] : 0.f) - mu) * inv * g[c] + b[c];
    o32[row * ncols + c] = v;
    if (o16) o16[row * ncols + c] = (_Float16)v;
  }
}

// W[K,N] f32 -> Wt[N,Kpad] f16 (zero pad K)
__global__ void k_wt(const float* __restrict__ Wsrc, _Float16* __restrict__ Wt, int K,
                     int N, int Kpad) {
  size_t i = (size_t)blockIdx.x * blockDim.x + threadIdx.x;
  if (i >= (size_t)N * Kpad) return;
  int kp = (int)(i % Kpad);
  int n  = (int)(i / Kpad);
  Wt[i] = (kp < K) ? (_Float16)Wsrc[(size_t)kp * N + n] : (_Float16)0.f;
}

// proj f32 [B,L,D] -> f16 [B,H,L,DH] (scaled)
__global__ void k_heads(const float* __restrict__ p, _Float16* __restrict__ o, float sc) {
  size_t i = (size_t)blockIdx.x * blockDim.x + threadIdx.x;
  if (i >= (size_t)B_ * H_ * L_ * DH_) return;
  int dh = (int)(i % DH_); size_t t = i / DH_;
  int l = (int)(t % L_); t /= L_;
  int h = (int)(t % H_); int b = (int)(t / H_);
  o[i] = (_Float16)(p[((size_t)b * L_ + l) * D_ + h * DH_ + dh] * sc);
}

// proj f32 [B,L,D] -> f16 [B,H,DH,L] (transposed per head)
__global__ void k_heads_t(const float* __restrict__ p, _Float16* __restrict__ o) {
  size_t i = (size_t)blockIdx.x * blockDim.x + threadIdx.x;
  if (i >= (size_t)B_ * H_ * DH_ * L_) return;
  int l = (int)(i % L_); size_t t = i / L_;
  int dh = (int)(t % DH_); t /= DH_;
  int h = (int)(t % H_); int b = (int)(t / H_);
  o[i] = (_Float16)p[((size_t)b * L_ + l) * D_ + h * DH_ + dh];
}

// extract first G tokens of qg proj -> f16 [B,H,G,DH] scaled
__global__ void k_qg(const float* __restrict__ p, _Float16* __restrict__ o, float sc) {
  size_t i = (size_t)blockIdx.x * blockDim.x + threadIdx.x;
  if (i >= (size_t)B_ * H_ * G_ * DH_) return;
  int dh = (int)(i % DH_); size_t t = i / DH_;
  int gg = (int)(t % G_); t /= G_;
  int h = (int)(t % H_); int b = (int)(t / H_);
  o[i] = (_Float16)(p[((size_t)b * L_ + gg) * D_ + h * DH_ + dh] * sc);
}

// attn f32 [B,H,L,DH] -> f16 [B,L,D]
__global__ void k_attn2ao(const float* __restrict__ a, _Float16* __restrict__ o) {
  size_t i = (size_t)blockIdx.x * blockDim.x + threadIdx.x;
  if (i >= (size_t)B_ * L_ * D_) return;
  int d = (int)(i % D_); size_t bl = i / D_;
  int l = (int)(bl % L_); int b = (int)(bl / L_);
  int h = d / DH_, dh = d % DH_;
  o[i] = (_Float16)a[(((size_t)(b * H_ + h)) * L_ + l) * DH_ + dh];
}

// ---------------- banded (local) attention, flash style ----------------
// grid: B*H*NC blocks, 512 threads (16 waves); wave w = q-tile w of the chunk
__global__ void k_band_attn(const _Float16* __restrict__ qh, const _Float16* __restrict__ kh,
                            const _Float16* __restrict__ vth,
                            const float* __restrict__ amask, const float* __restrict__ gmask,
                            float* __restrict__ attn) {
  __shared__ _Float16 plds[16][512];
  int blk = blockIdx.x;
  int c = blk % NC_;
  int h = (blk / NC_) % H_;
  int b = blk / (NC_ * H_);
  int wave = threadIdx.x >> 5;
  int lane = threadIdx.x & 31;
  int lan = lane & 15, hi = lane >> 4;
  const _Float16* Q  = qh  + ((size_t)(b * H_ + h)) * L_ * DH_;
  const _Float16* Kk = kh  + ((size_t)(b * H_ + h)) * L_ * DH_;
  const _Float16* Vt = vth + ((size_t)(b * H_ + h)) * DH_ * L_;
  int q0 = wave * 16;                       // q-tile base within chunk
  v16h a0 = frag_load_rowmajor(Q, DH_, c * W_ + q0, 0);
  v16h a1 = frag_load_rowmajor(Q, DH_, c * W_ + q0, 32);
  v8f o0 = {}, o1 = {}, o2 = {}, o3 = {};
  float m_run[8], l_run[8];
#pragma unroll
  for (int r = 0; r < 8; ++r) { m_run[r] = -3.0e38f; l_run[r] = 0.f; }

  for (int tt = 0; tt < 25; ++tt) {         // tile 0 = global G keys, 1..24 = band
    v8f s[2];
#pragma unroll
    for (int h2 = 0; h2 < 2; ++h2) {
      int key = (tt == 0) ? (h2 * 16 + lan)
                          : (c * W_ - W_ + (tt - 1) * 32 + h2 * 16 + lan);
      v16h b0 = frag_load_rowguard(Kk, DH_, key, L_, 0);
      v16h b1 = frag_load_rowguard(Kk, DH_, key, L_, 32);
      v8f ss = {};
      ss = wmma_f16(a0, b0, ss);
      ss = wmma_f16(a1, b1, ss);
#pragma unroll
      for (int r = 0; r < 8; ++r) {
        int ql = q0 + r + 8 * hi;           // query index within chunk
        bool ok;
        if (tt == 0) {
          ok = (h2 == 0) && (amask[(size_t)b * L_ + key] > 0.f);
        } else {
          int kj = (tt - 1) * 32 + h2 * 16 + lan;   // 0..3W-1
          int p  = c * W_ + kj - W_;                // absolute key position
          int dq = kj - ql - W_;
          ok = (dq <= W_) && (dq >= -W_) && (p >= 0) && (p < L_) &&
               (amask[(size_t)b * L_ + p] * (1.f - gmask[(size_t)b * L_ + p]) > 0.f);
        }
        ss[r] = ok ? ss[r] : NEGV;
      }
      s[h2] = ss;
    }
    // streaming softmax update (per-row over 32 keys; rows distributed r+8*hi)
#pragma unroll
    for (int r = 0; r < 8; ++r) {
      float mv = fmaxf(s[0][r], s[1][r]);
      for (int off = 1; off < 16; off <<= 1) mv = fmaxf(mv, __shfl_xor(mv, off, 32));
      float mnew = fmaxf(m_run[r], mv);
      float p0 = __expf(s[0][r] - mnew);
      float p1 = __expf(s[1][r] - mnew);
      s[0][r] = p0; s[1][r] = p1;
      float ls = p0 + p1;
      for (int off = 1; off < 16; off <<= 1) ls += __shfl_xor(ls, off, 32);
      float fac = __expf(m_run[r] - mnew);
      l_run[r] = l_run[r] * fac + ls;
      o0[r] *= fac; o1[r] *= fac; o2[r] *= fac; o3[r] *= fac;
      m_run[r] = mnew;
    }
    // stage P -> LDS -> A-fragment layout
#pragma unroll
    for (int r = 0; r < 8; ++r) {
      plds[wave][(r + 8 * hi) * 32 + lan]      = (_Float16)s[0][r];
      plds[wave][(r + 8 * hi) * 32 + 16 + lan] = (_Float16)s[1][r];
    }
    __syncthreads();
    v16h pf;
    {
      const _Float16* pl = &plds[wave][0];
#pragma unroll
      for (int e = 0; e < 16; ++e) {
        int j = e >> 1;
        int k = ((j >= 4) ? 16 : 0) + hi * 8 + (j & 3) * 2 + (e & 1);
        pf[e] = pl[lan * 32 + k];
      }
    }
    __syncthreads();
    // O += P * V  (Bt = Vt[DH, L], key columns guarded)
    int kb = (tt == 0) ? 0 : (c * W_ - W_ + (tt - 1) * 32);
    o0 = wmma_f16(pf, frag_load_colguard(Vt, L_, 0,  kb, L_), o0);
    o1 = wmma_f16(pf, frag_load_colguard(Vt, L_, 16, kb, L_), o1);
    o2 = wmma_f16(pf, frag_load_colguard(Vt, L_, 32, kb, L_), o2);
    o3 = wmma_f16(pf, frag_load_colguard(Vt, L_, 48, kb, L_), o3);
  }
#pragma unroll
  for (int r = 0; r < 8; ++r) {
    float linv = 1.f / fmaxf(l_run[r], 1e-30f);
    int qa = c * W_ + q0 + r + 8 * hi;
    float* dst = attn + (((size_t)(b * H_ + h)) * L_ + qa) * DH_;
    dst[lan]      = o0[r] * linv;
    dst[16 + lan] = o1[r] * linv;
    dst[32 + lan] = o2[r] * linv;
    dst[48 + lan] = o3[r] * linv;
  }
}

// ---------------- global-token full attention (G queries x L keys) ----------------
// grid: B*H blocks of 32 threads (1 wave); overwrites attn rows [0,G)
__global__ void k_gattn(const _Float16* __restrict__ qgh, const _Float16* __restrict__ kgh,
                        const _Float16* __restrict__ vgth, const float* __restrict__ amask,
                        float* __restrict__ attn) {
  __shared__ _Float16 plds[512];
  int h = blockIdx.x % H_;
  int b = blockIdx.x / H_;
  int lane = threadIdx.x & 31;
  int lan = lane & 15, hi = lane >> 4;
  const _Float16* Qg = qgh  + ((size_t)(b * H_ + h)) * G_ * DH_;
  const _Float16* Kg = kgh  + ((size_t)(b * H_ + h)) * L_ * DH_;
  const _Float16* Vt = vgth + ((size_t)(b * H_ + h)) * DH_ * L_;
  v16h a0 = frag_load_rowmajor(Qg, DH_, 0, 0);
  v16h a1 = frag_load_rowmajor(Qg, DH_, 0, 32);
  v8f o0 = {}, o1 = {}, o2 = {}, o3 = {};
  float m_run[8], l_run[8];
#pragma unroll
  for (int r = 0; r < 8; ++r) { m_run[r] = -3.0e38f; l_run[r] = 0.f; }
  for (int kt = 0; kt < L_ / 32; ++kt) {
    v8f s[2];
#pragma unroll
    for (int h2 = 0; h2 < 2; ++h2) {
      int key = kt * 32 + h2 * 16 + lan;
      v16h b0 = frag_load_rowguard(Kg, DH_, key, L_, 0);
      v16h b1 = frag_load_rowguard(Kg, DH_, key, L_, 32);
      v8f ss = {};
      ss = wmma_f16(a0, b0, ss);
      ss = wmma_f16(a1, b1, ss);
#pragma unroll
      for (int r = 0; r < 8; ++r)
        ss[r] = (amask[(size_t)b * L_ + key] > 0.f) ? ss[r] : NEGV;
      s[h2] = ss;
    }
#pragma unroll
    for (int r = 0; r < 8; ++r) {
      float mv = fmaxf(s[0][r], s[1][r]);
      for (int off = 1; off < 16; off <<= 1) mv = fmaxf(mv, __shfl_xor(mv, off, 32));
      float mnew = fmaxf(m_run[r], mv);
      float p0 = __expf(s[0][r] - mnew);
      float p1 = __expf(s[1][r] - mnew);
      s[0][r] = p0; s[1][r] = p1;
      float ls = p0 + p1;
      for (int off = 1; off < 16; off <<= 1) ls += __shfl_xor(ls, off, 32);
      float fac = __expf(m_run[r] - mnew);
      l_run[r] = l_run[r] * fac + ls;
      o0[r] *= fac; o1[r] *= fac; o2[r] *= fac; o3[r] *= fac;
      m_run[r] = mnew;
    }
#pragma unroll
    for (int r = 0; r < 8; ++r) {
      plds[(r + 8 * hi) * 32 + lan]      = (_Float16)s[0][r];
      plds[(r + 8 * hi) * 32 + 16 + lan] = (_Float16)s[1][r];
    }
    __syncthreads();
    v16h pf;
#pragma unroll
    for (int e = 0; e < 16; ++e) {
      int j = e >> 1;
      int k = ((j >= 4) ? 16 : 0) + hi * 8 + (j & 3) * 2 + (e & 1);
      pf[e] = plds[lan * 32 + k];
    }
    __syncthreads();
    int kb = kt * 32;
    o0 = wmma_f16(pf, frag_load_colguard(Vt, L_, 0,  kb, L_), o0);
    o1 = wmma_f16(pf, frag_load_colguard(Vt, L_, 16, kb, L_), o1);
    o2 = wmma_f16(pf, frag_load_colguard(Vt, L_, 32, kb, L_), o2);
    o3 = wmma_f16(pf, frag_load_colguard(Vt, L_, 48, kb, L_), o3);
  }
#pragma unroll
  for (int r = 0; r < 8; ++r) {
    float linv = 1.f / fmaxf(l_run[r], 1e-30f);
    int qa = r + 8 * hi;                       // rows 0..15 (global tokens)
    float* dst = attn + (((size_t)(b * H_ + h)) * L_ + qa) * DH_;
    dst[lan]      = o0[r] * linv;
    dst[16 + lan] = o1[r] * linv;
    dst[32 + lan] = o2[r] * linv;
    dst[48 + lan] = o3[r] * linv;
  }
}

// ---------------- GRU ----------------
// gin f16 [B*L, GIK_]: cols 0..767 = x, 768..777 = label_emb[mention], rest 0
__global__ void k_gin(const float* __restrict__ x, const int* __restrict__ ment,
                      const float* __restrict__ lemb, _Float16* __restrict__ gin) {
  size_t i = (size_t)blockIdx.x * blockDim.x + threadIdx.x;
  if (i >= (size_t)B_ * L_ * GIK_) return;
  int c = (int)(i % GIK_);
  size_t bl = i / GIK_;
  float v = 0.f;
  if (c < D_) v = x[bl * D_ + c];
  else if (c < D_ + LEMB_) v = lemb[(size_t)ment[bl] * LEMB_ + (c - D_)];
  gin[i] = (_Float16)v;
}

// sequential scan, one block (1024 threads) per direction
__global__ void k_gru(const _Float16* __restrict__ gi, const _Float16* __restrict__ whT,
                      const float* __restrict__ bh, float* __restrict__ hs, int reverse) {
  __shared__ float hbuf[B_][GRUH_];
  __shared__ float ghb[B_][3 * GRUH_];
  int tid = threadIdx.x;
  for (int i = tid; i < B_ * GRUH_; i += blockDim.x) hbuf[i / GRUH_][i % GRUH_] = 0.f;
  __syncthreads();
  for (int step = 0; step < L_; ++step) {
    int t = reverse ? (L_ - 1 - step) : step;
    for (int o = tid; o < B_ * 3 * GRUH_; o += blockDim.x) {
      int n = o % (3 * GRUH_);
      int b = o / (3 * GRUH_);
      const _Float16* wr = whT + (size_t)n * GRUH_;
      float acc = bh[n];
      for (int k = 0; k < GRUH_; ++k) acc += (float)wr[k] * hbuf[b][k];
      ghb[b][n] = acc;
    }
    __syncthreads();
    for (int o = tid; o < B_ * GRUH_; o += blockDim.x) {
      int j = o % GRUH_;
      int b = o / GRUH_;
      const _Float16* gir = gi + ((size_t)b * L_ + t) * 3 * GRUH_;
      float r = 1.f / (1.f + __expf(-((float)gir[j] + ghb[b][j])));
      float z = 1.f / (1.f + __expf(-((float)gir[GRUH_ + j] + ghb[b][GRUH_ + j])));
      float n = tanhf((float)gir[2 * GRUH_ + j] + r * ghb[b][2 * GRUH_ + j]);
      float hnew = (1.f - z) * n + z * hbuf[b][j];
      hs[((size_t)b * L_ + t) * GRUH_ + j] = hnew;
      hbuf[b][j] = hnew;
    }
    __syncthreads();
  }
}

// ---------------- classifier / loss ----------------
__global__ void k_cls(const float* __restrict__ hf, const float* __restrict__ hb,
                      const float* __restrict__ cw, const float* __restrict__ cb,
                      float* __restrict__ logits, float* __restrict__ preds) {
  size_t i = (size_t)blockIdx.x * blockDim.x + threadIdx.x;
  if (i >= (size_t)B_ * L_) return;
  const float* f = hf + i * GRUH_;
  const float* bk = hb + i * GRUH_;
  float lg[NLAB_];
#pragma unroll
  for (int c = 0; c < NLAB_; ++c) {
    float acc = cb[c];
    for (int k = 0; k < GRUH_; ++k) acc += f[k] * cw[(size_t)k * NLAB_ + c];
    for (int k = 0; k < GRUH_; ++k) acc += bk[k] * cw[(size_t)(GRUH_ + k) * NLAB_ + c];
    lg[c] = acc;
    logits[i * NLAB_ + c] = acc;
  }
  int am = 0;
  if (lg[1] > lg[am]) am = 1;
  if (lg[2] > lg[am]) am = 2;
  preds[i] = (float)am;
}

__global__ void k_zero(float* p, int n) {
  int i = blockIdx.x * blockDim.x + threadIdx.x;
  if (i < n) p[i] = 0.f;
}

__global__ void k_counts(const int* __restrict__ lab, const float* __restrict__ amask,
                         float* __restrict__ cnt) {
  size_t i = (size_t)blockIdx.x * blockDim.x + threadIdx.x;
  if (i >= (size_t)B_ * L_) return;
  float m = (amask[i] == 1.0f) ? 1.f : 0.f;
  if (m > 0.f) atomicAdd(&cnt[lab[i]], m);
}

__global__ void k_loss(const int* __restrict__ lab, const float* __restrict__ amask,
                       const float* __restrict__ logits, const float* __restrict__ cnt,
                       float* __restrict__ acc) {
  size_t i = (size_t)blockIdx.x * blockDim.x + threadIdx.x;
  if (i >= (size_t)B_ * L_) return;
  int la = lab[i];
  float m = (amask[i] == 1.0f) ? 1.f : 0.f;
  float w = m / (1.f + cnt[la]);
  const float* lg = logits + i * NLAB_;
  float mx = fmaxf(lg[0], fmaxf(lg[1], lg[2]));
  float se = __expf(lg[0] - mx) + __expf(lg[1] - mx) + __expf(lg[2] - mx);
  float nll = -(lg[la] - mx - logf(se));
  atomicAdd(&acc[0], w * nll);
  atomicAdd(&acc[1], w);
}

__global__ void k_final(const float* __restrict__ acc, float* __restrict__ out) {
  if (threadIdx.x == 0 && blockIdx.x == 0) out[0] = acc[0] / fmaxf(acc[1], 1e-30f);
}

// ---------------- host orchestration ----------------
static inline int eblocks(size_t n) { return (int)((n + 255) / 256); }

static void gemm(hipStream_t s, const _Float16* A, const _Float16* Bt, const float* bias,
                 float* c32, _Float16* c16, int M, int N, int K, int act) {
  int blocks = (M / BM_) * (N / BN_);
  k_gemm<<<blocks, 256, 0, s>>>(A, Bt, bias, c32, c16, M, N, K, act);
}

extern "C" void kernel_launch(void* const* d_in, const int* in_sizes, int n_in,
                              void* d_out, int out_size, void* d_ws, size_t ws_size,
                              hipStream_t stream) {
  // Input layout assumption: full jax pytree flatten (dict keys sorted at every level).
  const float* amask   = (const float*)d_in[0];
  const float* gmask   = (const float*)d_in[1];
  const int*   lab     = (const int*)d_in[2];
  const int*   ment    = (const int*)d_in[3];
  const float* cls_b   = (const float*)d_in[4];
  const float* cls_w   = (const float*)d_in[5];
  const float* eln_b   = (const float*)d_in[6];
  const float* eln_g   = (const float*)d_in[7];
  const float* grub_bh = (const float*)d_in[8];
  const float* grub_bi = (const float*)d_in[9];
  const float* grub_wh = (const float*)d_in[10];
  const float* grub_wi = (const float*)d_in[11];
  const float* gruf_bh = (const float*)d_in[12];
  const float* gruf_bi = (const float*)d_in[13];
  const float* gruf_wh = (const float*)d_in[14];
  const float* gruf_wi = (const float*)d_in[15];
  const float* lemb    = (const float*)d_in[16];
  // layers: 22 leaves each, base 17: ff1.b ff1.w ff2.b ff2.w k.b k.w kg.b kg.w
  //         ln1_b ln1_g ln2_b ln2_g o.b o.w q.b q.w qg.b qg.w v.b v.w vg.b vg.w
  const float* pemb    = (const float*)d_in[61];
  const float* temb    = (const float*)d_in[62];
  const int*   tok     = (const int*)d_in[63];

  // workspace bump allocator
  char* wp = (char*)d_ws;
  auto alloc = [&](size_t bytes) -> void* {
    void* p = (void*)wp;
    wp += (bytes + 255) & ~(size_t)255;
    return p;
  };
  const size_t SLD = (size_t)B_ * L_ * D_;            // 6.29M
  float*     xf32 = (float*)alloc(SLD * 4);
  _Float16*  xh   = (_Float16*)alloc(SLD * 2);
  float*     proj = (float*)alloc(SLD * 4);
  _Float16*  wt   = (_Float16*)alloc((size_t)FF_ * D_ * 2);   // largest Wt
  _Float16*  qh   = (_Float16*)alloc(SLD * 2);
  _Float16*  kh   = (_Float16*)alloc(SLD * 2);
  _Float16*  vth  = (_Float16*)alloc(SLD * 2);
  _Float16*  kgh  = (_Float16*)alloc(SLD * 2);
  _Float16*  vgth = (_Float16*)alloc(SLD * 2);
  _Float16*  qgh  = (_Float16*)alloc((size_t)B_ * H_ * G_ * DH_ * 2);
  float*     attn = (float*)alloc(SLD * 4);
  _Float16*  aoh  = (_Float16*)alloc(SLD * 2);
  _Float16*  ffh  = (_Float16*)alloc((size_t)B_ * L_ * FF_ * 2);
  _Float16*  ginh = (_Float16*)alloc((size_t)B_ * L_ * GIK_ * 2);
  _Float16*  gifh = (_Float16*)alloc((size_t)B_ * L_ * 3 * GRUH_ * 2);
  _Float16*  gibh = (_Float16*)alloc((size_t)B_ * L_ * 3 * GRUH_ * 2);
  _Float16*  whTf = (_Float16*)alloc((size_t)3 * GRUH_ * GRUH_ * 2);
  _Float16*  whTb = (_Float16*)alloc((size_t)3 * GRUH_ * GRUH_ * 2);
  float*     hf   = (float*)alloc((size_t)B_ * L_ * GRUH_ * 4);
  float*     hb   = (float*)alloc((size_t)B_ * L_ * GRUH_ * 4);
  float*     lgts = (float*)alloc((size_t)B_ * L_ * NLAB_ * 4);
  float*     cnt  = (float*)alloc(16 * 4);   // [0..2]=counts, [3]=pad
  float*     racc = (float*)alloc(16 * 4);   // [0]=sum w*nll, [1]=sum w
  (void)ws_size; (void)n_in; (void)in_sizes; (void)out_size;

  const int M = B_ * L_;
  const float qscale = 0.125f;  // DH^-0.5

  // ---- embedding + LN ----
  k_embed<<<eblocks(SLD), 256, 0, stream>>>(tok, temb, pemb, xf32);
  k_ln<<<M, 256, 0, stream>>>(xf32, nullptr, eln_g, eln_b, xf32, xh, D_);

  // ---- transformer layers ----
  for (int layer = 0; layer < 2; ++layer) {
    int base = 17 + 22 * layer;
    const float* ff1_b = (const float*)d_in[base + 0];
    const float* ff1_w = (const float*)d_in[base + 1];
    const float* ff2_b = (const float*)d_in[base + 2];
    const float* ff2_w = (const float*)d_in[base + 3];
    const float* k_b   = (const float*)d_in[base + 4];
    const float* k_w   = (const float*)d_in[base + 5];
    const float* kg_b  = (const float*)d_in[base + 6];
    const float* kg_w  = (const float*)d_in[base + 7];
    const float* ln1_b = (const float*)d_in[base + 8];
    const float* ln1_g = (const float*)d_in[base + 9];
    const float* ln2_b = (const float*)d_in[base + 10];
    const float* ln2_g = (const float*)d_in[base + 11];
    const float* o_b   = (const float*)d_in[base + 12];
    const float* o_w   = (const float*)d_in[base + 13];
    const float* q_b   = (const float*)d_in[base + 14];
    const float* q_w   = (const float*)d_in[base + 15];
    const float* qg_b  = (const float*)d_in[base + 16];
    const float* qg_w  = (const float*)d_in[base + 17];
    const float* v_b   = (const float*)d_in[base + 18];
    const float* v_w   = (const float*)d_in[base + 19];
    const float* vg_b  = (const float*)d_in[base + 20];
    const float* vg_w  = (const float*)d_in[base + 21];

    const size_t NW = (size_t)D_ * D_;
    // q
    k_wt<<<eblocks(NW), 256, 0, stream>>>(q_w, wt, D_, D_, D_);
    gemm(stream, xh, wt, q_b, proj, nullptr, M, D_, D_, 0);
    k_heads<<<eblocks(SLD), 256, 0, stream>>>(proj, qh, qscale);
    // k
    k_wt<<<eblocks(NW), 256, 0, stream>>>(k_w, wt, D_, D_, D_);
    gemm(stream, xh, wt, k_b, proj, nullptr, M, D_, D_, 0);
    k_heads<<<eblocks(SLD), 256, 0, stream>>>(proj, kh, 1.f);
    // v (transposed per head)
    k_wt<<<eblocks(NW), 256, 0, stream>>>(v_w, wt, D_, D_, D_);
    gemm(stream, xh, wt, v_b, proj, nullptr, M, D_, D_, 0);
    k_heads_t<<<eblocks(SLD), 256, 0, stream>>>(proj, vth);
    // kg (keys for global-token attention)
    k_wt<<<eblocks(NW), 256, 0, stream>>>(kg_w, wt, D_, D_, D_);
    gemm(stream, xh, wt, kg_b, proj, nullptr, M, D_, D_, 0);
    k_heads<<<eblocks(SLD), 256, 0, stream>>>(proj, kgh, 1.f);
    // vg (transposed)
    k_wt<<<eblocks(NW), 256, 0, stream>>>(vg_w, wt, D_, D_, D_);
    gemm(stream, xh, wt, vg_b, proj, nullptr, M, D_, D_, 0);
    k_heads_t<<<eblocks(SLD), 256, 0, stream>>>(proj, vgth);
    // qg (only first G rows kept)
    k_wt<<<eblocks(NW), 256, 0, stream>>>(qg_w, wt, D_, D_, D_);
    gemm(stream, xh, wt, qg_b, proj, nullptr, M, D_, D_, 0);
    k_qg<<<eblocks((size_t)B_ * H_ * G_ * DH_), 256, 0, stream>>>(proj, qgh, qscale);

    // attention
    k_band_attn<<<B_ * H_ * NC_, 512, 0, stream>>>(qh, kh, vth, amask, gmask, attn);
    k_gattn<<<B_ * H_, 32, 0, stream>>>(qgh, kgh, vgth, amask, attn);
    k_attn2ao<<<eblocks(SLD), 256, 0, stream>>>(attn, aoh);

    // output projection + residual LN1
    k_wt<<<eblocks(NW), 256, 0, stream>>>(o_w, wt, D_, D_, D_);
    gemm(stream, aoh, wt, o_b, proj, nullptr, M, D_, D_, 0);
    k_ln<<<M, 256, 0, stream>>>(proj, xf32, ln1_g, ln1_b, xf32, xh, D_);

    // FFN
    k_wt<<<eblocks((size_t)FF_ * D_), 256, 0, stream>>>(ff1_w, wt, D_, FF_, D_);
    gemm(stream, xh, wt, ff1_b, nullptr, ffh, M, FF_, D_, 1);   // fused GELU
    k_wt<<<eblocks((size_t)FF_ * D_), 256, 0, stream>>>(ff2_w, wt, FF_, D_, FF_);
    gemm(stream, ffh, wt, ff2_b, proj, nullptr, M, D_, FF_, 0);
    k_ln<<<M, 256, 0, stream>>>(proj, xf32, ln2_g, ln2_b, xf32, xh, D_);
  }

  // ---- GRU ----
  k_gin<<<eblocks((size_t)B_ * L_ * GIK_), 256, 0, stream>>>(xf32, ment, lemb, ginh);
  // input-side GEMMs (parallel over all timesteps)
  k_wt<<<eblocks((size_t)3 * GRUH_ * GIK_), 256, 0, stream>>>(gruf_wi, wt, D_ + LEMB_,
                                                              3 * GRUH_, GIK_);
  gemm(stream, ginh, wt, gruf_bi, nullptr, gifh, M, 3 * GRUH_, GIK_, 0);
  k_wt<<<eblocks((size_t)3 * GRUH_ * GIK_), 256, 0, stream>>>(grub_wi, wt, D_ + LEMB_,
                                                              3 * GRUH_, GIK_);
  gemm(stream, ginh, wt, grub_bi, nullptr, gibh, M, 3 * GRUH_, GIK_, 0);
  // recurrent weights
  k_wt<<<eblocks((size_t)3 * GRUH_ * GRUH_), 256, 0, stream>>>(gruf_wh, whTf, GRUH_,
                                                               3 * GRUH_, GRUH_);
  k_wt<<<eblocks((size_t)3 * GRUH_ * GRUH_), 256, 0, stream>>>(grub_wh, whTb, GRUH_,
                                                               3 * GRUH_, GRUH_);
  k_gru<<<1, 1024, 0, stream>>>(gifh, whTf, gruf_bh, hf, 0);
  k_gru<<<1, 1024, 0, stream>>>(gibh, whTb, grub_bh, hb, 1);

  // ---- classifier, predictions, loss ----
  float* preds = (float*)d_out;              // [B*L] predictions, [B*L] = loss
  k_cls<<<eblocks((size_t)B_ * L_), 256, 0, stream>>>(hf, hb, cls_w, cls_b, lgts, preds);
  k_zero<<<1, 32, 0, stream>>>(cnt, 4);
  k_zero<<<1, 32, 0, stream>>>(racc, 2);
  k_counts<<<eblocks((size_t)B_ * L_), 256, 0, stream>>>(lab, amask, cnt);
  k_loss<<<eblocks((size_t)B_ * L_), 256, 0, stream>>>(lab, amask, lgts, cnt, racc);
  k_final<<<1, 32, 0, stream>>>(racc, preds + (size_t)B_ * L_);
}
